// Attention_39633958208089
// MI455X (gfx1250) — matrix operand order repaired and checked by reading, not compiled, not run on hardware
//
#include <hip/hip_runtime.h>
#include <hip/hip_bf16.h>

// ---------------------------------------------------------------------------
// Problem constants (from reference): B=8, N=1024, C=256, H=8, D=64
//   INNER = H*D = 512, BH = B*H = 64, k1 = int(0.3*N) = 307
// Output: feature (B,C,N) f32 = 2,097,152 floats, then loc (B,307) as floats.
// ---------------------------------------------------------------------------

typedef __bf16 bf16_t;
typedef __attribute__((ext_vector_type(16))) __bf16 v16bf;
typedef __attribute__((ext_vector_type(8)))  __bf16 v8bf;
typedef __attribute__((ext_vector_type(8)))  float  v8f;
typedef __attribute__((ext_vector_type(4)))  unsigned int v4u;
typedef __attribute__((ext_vector_type(8)))  int v8i;
typedef __attribute__((ext_vector_type(4)))  int v4i;

#define K1SEL 307
#define SCALE 0.125f  // D^-0.5, D=64

__device__ inline v8f vzero8() {
  v8f z;
#pragma unroll
  for (int i = 0; i < 8; ++i) z[i] = 0.f;
  return z;
}

__device__ inline v8f wmma_bf16(v16bf a, v16bf b, v8f c) {
  // D = A(16x32 bf16) * B(32x16 bf16) + C(16x16 f32)
  return __builtin_amdgcn_wmma_f32_16x16x32_bf16(
      /*neg_a=*/false, a, /*neg_b=*/false, b,
      /*c_mod=*/(short)0, c, /*reuse_a=*/false, /*reuse_b=*/false);
}

// A operand (16x32, MxK). Lanes 0-15 = rows M, halves 0..7 = K 0..7,
// halves 8..15 = K 16..23; lanes 16-31 shift K-base by 8 (ISA A layout).
__device__ inline v16bf load_a_bf16(const bf16_t* p, int ld, int lane) {
  const int m  = lane & 15;
  const int kb = (lane & 16) ? 8 : 0;
  const bf16_t* r = p + (size_t)m * ld + kb;
  v8bf lo = *(const v8bf*)(r);
  v8bf hi = *(const v8bf*)(r + 16);
  v16bf a;
#pragma unroll
  for (int i = 0; i < 8; ++i) { a[i] = lo[i]; a[i + 8] = hi[i]; }
  return a;
}

// B operand (32x16, KxN). Lane n holds column n; halves 0..7 = K 0..7,
// halves 8..15 = K 16..23; lanes 16-31 shift K-base by 8. K contiguous with row stride `ld`.
__device__ inline v16bf load_b_bf16(const bf16_t* p, int ld, int lane) {
  const int n  = lane & 15;
  const int kb = (lane & 16) ? 8 : 0;
  const bf16_t* r = p + (size_t)n * ld + kb;
  v8bf lo = *(const v8bf*)(r);
  v8bf hi = *(const v8bf*)(r + 16);
  v16bf b;
#pragma unroll
  for (int i = 0; i < 8; ++i) { b[i] = lo[i]; b[i + 8] = hi[i]; }
  return b;
}

#if __has_builtin(__builtin_amdgcn_tensor_load_to_lds)
#define HAVE_TDM 1
// Tensor Data Mover: 2-D tile (16-bit elements) from global to LDS, packed.
// D# bitfields per CDNA5 ISA ch.8: group0 = {count|flags, lds_addr,
// global_addr lo, global_addr hi | type=2}; group1 = dims/strides. This
// toolchain exposes the 6-operand builtin (4 descriptor groups + extra
// int32x8 + cpol); groups beyond 1 are unused for <=2D tiles -> zeros.
__device__ inline void tdm_load_2d_bf16(unsigned lds_off, const void* gptr,
                                        unsigned tile_x, unsigned tile_y,
                                        unsigned tensor_x, unsigned tensor_y,
                                        unsigned stride_x /*elements*/) {
  unsigned long long ga = (unsigned long long)(size_t)gptr;
  v4u g0;
  g0[0] = 1u;                                        // count=1 (user mode)
  g0[1] = lds_off;                                   // lds_addr (bytes)
  g0[2] = (unsigned)(ga & 0xFFFFFFFFu);              // global_addr[31:0]
  g0[3] = (unsigned)((ga >> 32) & 0x01FFFFFFu) | (2u << 30);  // [56:32]|type=2
  v8i g1;
  g1[0] = (int)(1u << 16);                           // wg_mask=0, data_size=2B
  g1[1] = (int)((tensor_x & 0xFFFFu) << 16);         // tensor_dim0[15:0]@48
  g1[2] = (int)(((tensor_x >> 16) & 0xFFFFu) |       // tensor_dim0[31:16]
                ((tensor_y & 0xFFFFu) << 16));       // tensor_dim1[15:0]
  g1[3] = (int)(((tensor_y >> 16) & 0xFFFFu) |       // tensor_dim1[31:16]
                ((tile_x & 0xFFFFu) << 16));         // tile_dim0
  g1[4] = (int)(tile_y & 0xFFFFu);                   // tile_dim1 (tile_dim2=0)
  g1[5] = (int)stride_x;                             // tensor_dim0_stride lo
  g1[6] = 0;                                         // stride0 hi | stride1 lo
  g1[7] = 0;                                         // stride1 hi
  v4i gz4 = {0, 0, 0, 0};
  v8i gz8 = {0, 0, 0, 0, 0, 0, 0, 0};
  __builtin_amdgcn_tensor_load_to_lds(g0, g1, gz4, gz4, gz8, 0);
}
#else
#define HAVE_TDM 0
#endif

// ---------------------------------------------------------------------------
// K0: f32 -> bf16 conversion (x, Wq, Wk, Wv, Wo staged once in workspace)
// ---------------------------------------------------------------------------
__global__ void k_f32_to_bf16(const float* __restrict__ src,
                              bf16_t* __restrict__ dst, int n) {
  int i = blockIdx.x * blockDim.x + threadIdx.x;
  if (i < n) dst[i] = (bf16_t)src[i];
}

// ---------------------------------------------------------------------------
// K1: QKV projection. One wave per 16x64 strip (4 N-tiles share one A
// fragment per k-step -> 4x less A traffic). Q,K stored (BH, N, 64) with D
// contiguous; V stored transposed (BH, 64, N) so P*V B-operand is contiguous.
// ---------------------------------------------------------------------------
__global__ void k_qkv(const bf16_t* __restrict__ xbf,
                      const bf16_t* __restrict__ wq,
                      const bf16_t* __restrict__ wk,
                      const bf16_t* __restrict__ wv,
                      bf16_t* __restrict__ Qb, bf16_t* __restrict__ Kb,
                      bf16_t* __restrict__ Vt) {
  const int lane = threadIdx.x & 31;
  const int wid  = (int)((blockIdx.x * blockDim.x + threadIdx.x) >> 5); // 0..12287
  const int mat  = wid / 4096;             // 0:Q 1:K 2:V
  const int rem  = wid - mat * 4096;
  const int mt   = rem >> 3;               // 0..511 (rows of B*N in 16s)
  const int ntg  = rem & 7;                // 0..7   (groups of 4 col-tiles)
  const bf16_t* W = (mat == 0) ? wq : (mat == 1) ? wk : wv;
  const bf16_t* Arow = xbf + (size_t)mt * 16 * 256;
  v8f acc[4];
#pragma unroll
  for (int t = 0; t < 4; ++t) acc[t] = vzero8();
#pragma unroll
  for (int kk = 0; kk < 256; kk += 32) {
    v16bf a = load_a_bf16(Arow + kk, 256, lane);
#pragma unroll
    for (int t = 0; t < 4; ++t) {
      const bf16_t* Brow = W + (size_t)(ntg * 4 + t) * 16 * 256;
      acc[t] = wmma_bf16(a, load_b_bf16(Brow + kk, 256, lane), acc[t]);
    }
  }
  const int rhalf = (lane >> 4) & 1;
  const int nloc  = lane & 15;
#pragma unroll
  for (int t = 0; t < 4; ++t) {
    const int ncol = (ntg * 4 + t) * 16 + nloc;
    const int h = ncol >> 6, d = ncol & 63;
#pragma unroll
    for (int v = 0; v < 8; ++v) {
      int m   = mt * 16 + v + 8 * rhalf;
      int b_  = m >> 10, tok = m & 1023;
      int bh  = b_ * 8 + h;
      bf16_t val = (bf16_t)acc[t][v];
      if (mat == 0)      Qb[((size_t)bh * 1024 + tok) * 64 + d] = val;
      else if (mat == 1) Kb[((size_t)bh * 1024 + tok) * 64 + d] = val;
      else               Vt[((size_t)bh * 64 + d) * 1024 + tok] = val;
    }
  }
}

// ---------------------------------------------------------------------------
// K2: per-row softmax stats (online max/sum over all 1024 columns).
// ---------------------------------------------------------------------------
__global__ void k_rowstats(const bf16_t* __restrict__ Qb,
                           const bf16_t* __restrict__ Kb,
                           float* __restrict__ rowmax,
                           float* __restrict__ rowinv) {
  const int lane = threadIdx.x & 31;
  const int wid  = (int)((blockIdx.x * blockDim.x + threadIdx.x) >> 5); // 0..4095
  const int bh = wid >> 6, mt = wid & 63;
  const bf16_t* Arow = Qb + ((size_t)bh * 1024 + (size_t)mt * 16) * 64;
  v16bf a0 = load_a_bf16(Arow + 0,  64, lane);
  v16bf a1 = load_a_bf16(Arow + 32, 64, lane);
  float mx[8], sm[8];
#pragma unroll
  for (int v = 0; v < 8; ++v) { mx[v] = -1e30f; sm[v] = 0.f; }
  for (int nt = 0; nt < 64; ++nt) {
    const bf16_t* Brow = Kb + ((size_t)bh * 1024 + (size_t)nt * 16) * 64;
    v8f c = vzero8();
    c = wmma_bf16(a0, load_b_bf16(Brow + 0,  64, lane), c);
    c = wmma_bf16(a1, load_b_bf16(Brow + 32, 64, lane), c);
#pragma unroll
    for (int v = 0; v < 8; ++v) {
      float s  = c[v] * SCALE;
      float nm = fmaxf(mx[v], s);
      sm[v] = sm[v] * __expf(mx[v] - nm) + __expf(s - nm);
      mx[v] = nm;
    }
  }
#pragma unroll
  for (int xm = 1; xm < 16; xm <<= 1) {
#pragma unroll
    for (int v = 0; v < 8; ++v) {
      float om = __shfl_xor(mx[v], xm, 32);
      float os = __shfl_xor(sm[v], xm, 32);
      float nm = fmaxf(mx[v], om);
      sm[v] = sm[v] * __expf(mx[v] - nm) + os * __expf(om - nm);
      mx[v] = nm;
    }
  }
  if ((lane & 15) == 0) {
    int base = bh * 1024 + mt * 16 + 8 * (lane >> 4);
#pragma unroll
    for (int v = 0; v < 8; ++v) {
      rowmax[base + v] = mx[v];
      rowinv[base + v] = 1.0f / sm[v];
    }
  }
}

// ---------------------------------------------------------------------------
// K3: column scores. col_score[bh,n] = (1/N) * sum_m attn[bh,m,n].
// ---------------------------------------------------------------------------
__global__ void k_colscore(const bf16_t* __restrict__ Qb,
                           const bf16_t* __restrict__ Kb,
                           const float* __restrict__ rowmax,
                           const float* __restrict__ rowinv,
                           float* __restrict__ colscore) {
  const int lane = threadIdx.x & 31;
  const int wid  = (int)((blockIdx.x * blockDim.x + threadIdx.x) >> 5); // 0..4095
  const int bh = wid >> 6, nt = wid & 63;
  const bf16_t* Brow = Kb + ((size_t)bh * 1024 + (size_t)nt * 16) * 64;
  v16bf b0 = load_b_bf16(Brow + 0,  64, lane);
  v16bf b1 = load_b_bf16(Brow + 32, 64, lane);
  const int rhalf = (lane >> 4) & 1;
  float colacc = 0.f;
  for (int mt = 0; mt < 64; ++mt) {
    const bf16_t* Arow = Qb + ((size_t)bh * 1024 + (size_t)mt * 16) * 64;
    v8f c = vzero8();
    c = wmma_bf16(load_a_bf16(Arow + 0,  64, lane), b0, c);
    c = wmma_bf16(load_a_bf16(Arow + 32, 64, lane), b1, c);
    int rbase = bh * 1024 + mt * 16 + 8 * rhalf;
#pragma unroll
    for (int v = 0; v < 8; ++v) {
      colacc += __expf(c[v] * SCALE - rowmax[rbase + v]) * rowinv[rbase + v];
    }
  }
  colacc += __shfl_xor(colacc, 16, 32);
  if (lane < 16) colscore[bh * 1024 + nt * 16 + lane] = colacc * (1.0f / 1024.0f);
}

// ---------------------------------------------------------------------------
// K4: exact top-k (jax tie order: value desc, index asc) by rank counting.
// ---------------------------------------------------------------------------
__global__ void k_coltopk(const float* __restrict__ colscore,
                          float* __restrict__ colmask) {
  __shared__ float s[1024];
  const int bh = blockIdx.x;
  for (int i = threadIdx.x; i < 1024; i += 256) s[i] = colscore[bh * 1024 + i];
  __syncthreads();
  for (int i = threadIdx.x; i < 1024; i += 256) {
    float si = s[i];
    int cnt = 0;
    for (int j = 0; j < 1024; ++j) {
      float sj = s[j];
      cnt += ((sj > si) || (sj == si && j < i)) ? 1 : 0;
    }
    colmask[bh * 1024 + i] = (cnt < K1SEL) ? 1.0f : 0.0f;
  }
}

// ---------------------------------------------------------------------------
// K5: fused masked attention * V. All 8 waves of a block share the same K/V
// column chunks, so the block stages a 4KB K-chunk + 4KB V-chunk in LDS once
// per iteration (via the Tensor Data Mover on toolchains that expose it, else
// cooperative b128 copies), then every wave feeds its WMMA B operands from
// LDS -> 8x less repeated global traffic. Masked P tiles are staged per-wave
// in LDS (bf16, D-layout -> A-layout). Deterministic per-strip column sums of
// sparse attn are written for `loc`. row_score == 1/N in exact math, so the
// row top-k is rows 0..306 (jax stable tie-break).
// ---------------------------------------------------------------------------
__global__ void k_out(const bf16_t* __restrict__ Qb,
                      const bf16_t* __restrict__ Kb,
                      const bf16_t* __restrict__ Vt,
                      const float* __restrict__ rowmax,
                      const float* __restrict__ rowinv,
                      const float* __restrict__ colmask,
                      float* __restrict__ Apart,
                      bf16_t* __restrict__ outbf) {
  __shared__ __align__(16) bf16_t kst[32][64];     // [token_local][d]   4KB
  __shared__ __align__(16) bf16_t vst[64][32];     // [d][token_local]   4KB
  __shared__ __align__(16) bf16_t pst[8][16][32];  // per-wave masked P  8KB
  const int lane  = threadIdx.x & 31;
  const int w     = threadIdx.x >> 5;     // wave in block (8 waves)
  const int bh    = blockIdx.x >> 3;
  const int rb    = blockIdx.x & 7;
  const int strip = rb * 8 + w;           // 0..63 within bh
  const int row0  = strip * 16;
  const int rhalf = (lane >> 4) & 1;
  const int nloc  = lane & 15;
  const bf16_t* Arow = Qb + ((size_t)bh * 1024 + row0) * 64;
  v16bf aq0 = load_a_bf16(Arow + 0,  64, lane);
  v16bf aq1 = load_a_bf16(Arow + 32, 64, lane);

  float rm[8], ri[8], rmask[8];
  {
    int rbase = bh * 1024 + row0 + 8 * rhalf;
#pragma unroll
    for (int v = 0; v < 8; ++v) {
      rm[v] = rowmax[rbase + v];
      ri[v] = rowinv[rbase + v];
      rmask[v] = ((row0 + v + 8 * rhalf) < K1SEL) ? 1.0f : 0.0f;
    }
  }
  v8f acc[4];
#pragma unroll
  for (int j = 0; j < 4; ++j) acc[j] = vzero8();

  for (int ch = 0; ch < 32; ++ch) {       // 32 chunks of 32 columns
    const int tok0 = ch * 32;
    // ---- stage K chunk (32 tok x 64 d, contiguous) and V chunk (64 d x 32
    //      tok, row stride 1024) into LDS --------------------------------
#if HAVE_TDM
    if (threadIdx.x == 0) {
      tdm_load_2d_bf16((unsigned)(size_t)&kst[0][0],
                       Kb + ((size_t)bh * 1024 + tok0) * 64,
                       /*tile_x=*/2048, /*tile_y=*/1,
                       /*tensor_x=*/2048, /*tensor_y=*/1, /*stride_x=*/2048);
      tdm_load_2d_bf16((unsigned)(size_t)&vst[0][0],
                       Vt + ((size_t)bh * 64) * 1024 + tok0,
                       /*tile_x=*/32, /*tile_y=*/64,
                       /*tensor_x=*/1024, /*tensor_y=*/64, /*stride_x=*/1024);
      __builtin_amdgcn_s_wait_tensorcnt(0);
    }
#else
    {
      const int tid = threadIdx.x;                       // 256 threads
      ((v8bf*)&kst[0][0])[tid] =
          *(const v8bf*)(Kb + ((size_t)bh * 1024 + tok0) * 64 + tid * 8);
      const int vy = tid >> 2, vx = (tid & 3) * 8;
      ((v8bf*)&vst[vy][0])[tid & 3] =
          *(const v8bf*)(Vt + ((size_t)bh * 64 + vy) * 1024 + tok0 + vx);
    }
#endif
    __syncthreads();

    // ---- sim -> masked attention (P) + column partial sums -------------
#pragma unroll
    for (int j = 0; j < 2; ++j) {
      const int n0 = tok0 + j * 16;
      v8f c = vzero8();
      c = wmma_bf16(aq0, load_b_bf16(&kst[j * 16][0], 64, lane), c);
      c = wmma_bf16(aq1, load_b_bf16(&kst[j * 16][32], 64, lane), c);
      const int col = n0 + nloc;
      const float cm = colmask[bh * 1024 + col];
      float colpart = 0.f;
#pragma unroll
      for (int v = 0; v < 8; ++v) {
        float p  = __expf(c[v] * SCALE - rm[v]) * ri[v];
        float pm = p * fmaxf(rmask[v], cm);     // sparse_attn element
        colpart += pm;
        pst[w][v + 8 * rhalf][j * 16 + nloc] = (bf16_t)pm;
      }
      colpart += __shfl_xor(colpart, 16, 32);
      if (lane < 16)
        Apart[((size_t)(bh * 64 + strip)) * 1024 + col] = colpart;
    }
    // ---- P * V (B operand from LDS-staged, transposed V) ---------------
    v16bf aP = load_a_bf16(&pst[w][0][0], 32, lane);
#pragma unroll
    for (int j2 = 0; j2 < 4; ++j2) {
      acc[j2] = wmma_bf16(aP, load_b_bf16(&vst[j2 * 16][0], 32, lane), acc[j2]);
    }
    __syncthreads();   // protect kst/vst against next iteration's staging
  }
  // store out tile (16 tokens x 64 dims) as bf16 into (B*N, INNER)
  const int b_ = bh >> 3, h = bh & 7;
#pragma unroll
  for (int j2 = 0; j2 < 4; ++j2) {
#pragma unroll
    for (int v = 0; v < 8; ++v) {
      int tok  = row0 + v + 8 * rhalf;
      int colI = h * 64 + j2 * 16 + nloc;
      outbf[((size_t)(b_ * 1024 + tok)) * 512 + colI] = (bf16_t)acc[j2][v];
    }
  }
}

// ---------------------------------------------------------------------------
// K6: loc = topk over A_mean (fixed-order reduction of per-strip partials,
// then exact rank-counting topk). Writes indices as floats after feature.
// ---------------------------------------------------------------------------
__global__ void k_loc(const float* __restrict__ Apart,
                      float* __restrict__ locout) {
  __shared__ float s[1024];
  const int b_ = blockIdx.x;
  for (int i = threadIdx.x; i < 1024; i += 256) {
    float sum = 0.f;
    for (int h = 0; h < 8; ++h) {
      const float* base = Apart + ((size_t)((b_ * 8 + h) * 64)) * 1024 + i;
      for (int st = 0; st < 64; ++st) sum += base[(size_t)st * 1024];
    }
    s[i] = sum * (1.0f / (8.0f * 1024.0f));
  }
  __syncthreads();
  for (int i = threadIdx.x; i < 1024; i += 256) {
    float si = s[i];
    int cnt = 0;
    for (int j = 0; j < 1024; ++j) {
      float sj = s[j];
      cnt += ((sj > si) || (sj == si && j < i)) ? 1 : 0;
    }
    if (cnt < K1SEL) locout[b_ * K1SEL + cnt] = (float)i;
  }
}

// ---------------------------------------------------------------------------
// K7: feature = out @ Wo^T + bo, stored transposed (B, C, N).
// One wave per 16x64 strip (4 N-tiles reuse each A fragment).
// ---------------------------------------------------------------------------
__global__ void k_final(const bf16_t* __restrict__ outbf,
                        const bf16_t* __restrict__ wo,
                        const float* __restrict__ bo,
                        float* __restrict__ feature) {
  const int lane = threadIdx.x & 31;
  const int wid  = (int)((blockIdx.x * blockDim.x + threadIdx.x) >> 5); // 0..2047
  const int mt = wid >> 2, ntg = wid & 3;
  const bf16_t* Arow = outbf + (size_t)mt * 16 * 512;
  v8f acc[4];
#pragma unroll
  for (int t = 0; t < 4; ++t) acc[t] = vzero8();
  for (int kk = 0; kk < 512; kk += 32) {
    v16bf a = load_a_bf16(Arow + kk, 512, lane);
#pragma unroll
    for (int t = 0; t < 4; ++t) {
      const bf16_t* Brow = wo + (size_t)(ntg * 4 + t) * 16 * 512;
      acc[t] = wmma_bf16(a, load_b_bf16(Brow + kk, 512, lane), acc[t]);
    }
  }
  const int rhalf = (lane >> 4) & 1;
#pragma unroll
  for (int t = 0; t < 4; ++t) {
    const int cidx = (ntg * 4 + t) * 16 + (lane & 15);
    const float bias = bo[cidx];
#pragma unroll
    for (int v = 0; v < 8; ++v) {
      int m = mt * 16 + v + 8 * rhalf;
      int b_ = m >> 10, tok = m & 1023;
      feature[((size_t)(b_ * 256 + cidx)) * 1024 + tok] = acc[t][v] + bias;
    }
  }
}

// ---------------------------------------------------------------------------
// Host-side launcher
// ---------------------------------------------------------------------------
extern "C" void kernel_launch(void* const* d_in, const int* in_sizes, int n_in,
                              void* d_out, int out_size, void* d_ws, size_t ws_size,
                              hipStream_t stream) {
  (void)in_sizes; (void)n_in; (void)out_size; (void)ws_size;
  const float* x  = (const float*)d_in[0];  // (8,1024,256)
  const float* Wq = (const float*)d_in[1];  // (512,256)
  const float* Wk = (const float*)d_in[2];
  const float* Wv = (const float*)d_in[3];
  const float* Wo = (const float*)d_in[4];  // (256,512)
  const float* bo = (const float*)d_in[5];  // (256,)

  char* ws = (char*)d_ws;
  const size_t szXbf   = (size_t)8 * 1024 * 256 * 2;    // 4 MB
  const size_t szW     = (size_t)512 * 256 * 2;         // 256 KB each
  const size_t szQKV   = (size_t)64 * 1024 * 64 * 2;    // 8 MB each
  const size_t szRow   = (size_t)64 * 1024 * 4;         // 256 KB each
  const size_t szApart = (size_t)64 * 64 * 1024 * 4;    // 16 MB
  const size_t szOutbf = (size_t)8 * 1024 * 512 * 2;    // 8 MB

  bf16_t* xbf   = (bf16_t*)(ws);                 ws += szXbf;
  bf16_t* wq_b  = (bf16_t*)(ws);                 ws += szW;
  bf16_t* wk_b  = (bf16_t*)(ws);                 ws += szW;
  bf16_t* wv_b  = (bf16_t*)(ws);                 ws += szW;
  bf16_t* wo_b  = (bf16_t*)(ws);                 ws += szW;
  bf16_t* Qb    = (bf16_t*)(ws);                 ws += szQKV;
  bf16_t* Kb    = (bf16_t*)(ws);                 ws += szQKV;
  bf16_t* Vt    = (bf16_t*)(ws);                 ws += szQKV;
  float*  rowmax   = (float*)(ws);               ws += szRow;
  float*  rowinv   = (float*)(ws);               ws += szRow;
  float*  colscore = (float*)(ws);               ws += szRow;
  float*  colmask  = (float*)(ws);               ws += szRow;
  float*  Apart    = (float*)(ws);               ws += szApart;
  bf16_t* outbf    = (bf16_t*)(ws);              ws += szOutbf;

  float* feature = (float*)d_out;                            // 2,097,152 floats
  float* locout  = (float*)d_out + (size_t)8 * 256 * 1024;   // 8*307 floats

  // K0: stage bf16 copies
  k_f32_to_bf16<<<(2097152 + 255) / 256, 256, 0, stream>>>(x,  xbf,  2097152);
  k_f32_to_bf16<<<(131072 + 255) / 256, 256, 0, stream>>>(Wq, wq_b, 131072);
  k_f32_to_bf16<<<(131072 + 255) / 256, 256, 0, stream>>>(Wk, wk_b, 131072);
  k_f32_to_bf16<<<(131072 + 255) / 256, 256, 0, stream>>>(Wv, wv_b, 131072);
  k_f32_to_bf16<<<(131072 + 255) / 256, 256, 0, stream>>>(Wo, wo_b, 131072);

  // K1: QKV projection (12288 waves, 8 waves/block, 4-tile N blocking)
  k_qkv<<<1536, 256, 0, stream>>>(xbf, wq_b, wk_b, wv_b, Qb, Kb, Vt);
  // K2: row softmax stats (4096 waves)
  k_rowstats<<<512, 256, 0, stream>>>(Qb, Kb, rowmax, rowinv);
  // K3: column scores (4096 waves)
  k_colscore<<<512, 256, 0, stream>>>(Qb, Kb, rowmax, rowinv, colscore);
  // K4: column top-k mask
  k_coltopk<<<64, 256, 0, stream>>>(colscore, colmask);
  // K5: fused masked attention * V (TDM/LDS-staged K,V chunks)
  k_out<<<512, 256, 0, stream>>>(Qb, Kb, Vt, rowmax, rowinv, colmask,
                                 Apart, outbf);
  // K6: loc top-k
  k_loc<<<8, 256, 0, stream>>>(Apart, locout);
  // K7: output projection + transpose + bias (4-tile N blocking)
  k_final<<<256, 256, 0, stream>>>(outbf, wo_b, bo, feature);
}
